// Attention_83983790506152
// MI455X (gfx1250) — matrix operand order repaired and checked by reading, not compiled, not run on hardware
//
#include <hip/hip_runtime.h>
#include <hip/hip_bf16.h>

// ---------------------------------------------------------------------------
// Problem constants (from reference): B=4, N=2048, D=1024, H=16, d=64
// ---------------------------------------------------------------------------
static constexpr int BATCH   = 4;
static constexpr int SEQ     = 2048;
static constexpr int DIM     = 1024;
static constexpr int HEADS   = 16;
static constexpr int HD      = 64;          // head dim
static constexpr int D3      = 3 * DIM;     // 3072
static constexpr int MROWS   = BATCH * SEQ; // 8192
static constexpr float SCALE  = 0.125f;     // 64^-0.5
static constexpr float NEGINF = -1e30f;

typedef __attribute__((ext_vector_type(16))) __bf16 v16bf;
typedef __attribute__((ext_vector_type(8)))  float  v8f;
typedef __attribute__((ext_vector_type(4)))  int    v4i;

// ---------------------------------------------------------------------------
// CDNA5 async global->LDS copy (ASYNCcnt-tracked), guarded so the file still
// compiles on toolchains without the builtin.
// ---------------------------------------------------------------------------
#if defined(__HIP_DEVICE_COMPILE__) && \
    __has_builtin(__builtin_amdgcn_global_load_async_to_lds_b128)
#define USE_ASYNC_LDS 1
#else
#define USE_ASYNC_LDS 0
#endif

// Copy 16 bytes global -> LDS for this lane (async when available).
__device__ __forceinline__ void g2l_copy16(__bf16* lds_dst, const __bf16* gsrc) {
#if USE_ASYNC_LDS
  __builtin_amdgcn_global_load_async_to_lds_b128((v4i*)gsrc, (v4i*)lds_dst, 0, 0);
#else
  *(uint4*)lds_dst = *(const uint4*)gsrc;
#endif
}

__device__ __forceinline__ void async_wait0() {
#if USE_ASYNC_LDS
#if __has_builtin(__builtin_amdgcn_s_wait_asynccnt)
  __builtin_amdgcn_s_wait_asynccnt(0);
#else
  asm volatile("s_wait_asynccnt 0x0" ::: "memory");
#endif
#endif
}

// Stage 8 consecutive elements to LDS as bf16. f32 source: load+convert.
// bf16 source: straight (async) 16B copy.
template <typename T>
__device__ __forceinline__ void stage8(__bf16* dst, const T* src) {
  if constexpr (sizeof(T) == 4) {
    float4 a = *(const float4*)src;
    float4 b = *(const float4*)(src + 4);
    __bf16 t[8] = {(__bf16)a.x, (__bf16)a.y, (__bf16)a.z, (__bf16)a.w,
                   (__bf16)b.x, (__bf16)b.y, (__bf16)b.z, (__bf16)b.w};
    *(uint4*)dst = *(const uint4*)t;
  } else {
    g2l_copy16(dst, (const __bf16*)src);
  }
}

// ---------------------------------------------------------------------------
// WMMA helpers (CDNA5 wave32, D = A(16x32) * B(32x16) + C, f32 accum)
// ---------------------------------------------------------------------------
__device__ __forceinline__ v8f wmma_bf16(v16bf a, v16bf b, v8f c) {
  return __builtin_amdgcn_wmma_f32_16x16x32_bf16(false, a, false, b,
                                                 (short)0, c, false, false);
}

// A fragment (16x32 bf16) from row-major tile at p, leading dim ld.
// ISA: lanes 0-15 -> K 0..7 / 16..23 ; lanes 16-31 -> K 8..15 / 24..31
__device__ __forceinline__ v16bf lds_a_16x32(const __bf16* p, int ld, int lane) {
  int m  = lane & 15;
  int kb = (lane >> 4) << 3;
  v16bf a;
#pragma unroll
  for (int j = 0; j < 8; ++j) {
    a[j]     = p[m * ld + kb + j];
    a[j + 8] = p[m * ld + 16 + kb + j];
  }
  return a;
}

// B fragment (32x16 bf16) from row-major [K][N] tile at p, leading dim ld.
// ISA: lane = column n; lanes 0-15 hold K 0..7 / 16..23, lanes 16-31 hold K 8..15 / 24..31.
__device__ __forceinline__ v16bf lds_b_32x16(const __bf16* p, int ld, int lane) {
  int n  = lane & 15;
  int kb = (lane >> 4) << 3;
  v16bf b;
#pragma unroll
  for (int j = 0; j < 16; ++j) b[j] = p[(kb + (j & 7) + ((j >> 3) << 4)) * ld + n];
  return b;
}

// B fragment (32x16) = X^T, X row-major [16 rows = n][cols = k] at p.
// Used for S = Q * K^T (reads contiguous in k).
__device__ __forceinline__ v16bf lds_bT_16rows(const __bf16* p, int ld, int lane) {
  int n  = lane & 15;
  int kb = (lane >> 4) << 3;
  v16bf b;
#pragma unroll
  for (int j = 0; j < 16; ++j) b[j] = p[n * ld + kb + (j & 7) + ((j >> 3) << 4)];
  return b;
}

// ---------------------------------------------------------------------------
// Tiled GEMM: C[M][Nc] = A[M][K] * Bm[K][Nc]; 64x64 output tile, BK=64,
// 128-thread block (4 waves, one 16x64 strip each), 8 WMMAs per K step.
// Source operands may be f32 (converted while staging) or bf16 (async copy).
// BF16_OUT: store bf16 (QKV). Else: f32 + bias (projection -> d_out).
// ---------------------------------------------------------------------------
template <typename TA, typename TB, bool BF16_OUT>
__global__ __launch_bounds__(128) void gemm64_kernel(
    const TA* __restrict__ A, const TB* __restrict__ Bm,
    void* __restrict__ Cv, const float* __restrict__ bias, int K, int Nc) {
  __shared__ __align__(16) __bf16 As[64 * 64];
  __shared__ __align__(16) __bf16 Bs[64 * 64];

  const int tid  = threadIdx.x;
  const int wv   = tid >> 5;
  const int lane = tid & 31;
  const int nl   = lane & 15;
  const int mb   = (lane >> 4) << 3;
  const int m0   = blockIdx.y * 64;
  const int n0   = blockIdx.x * 64;

  v8f acc[4] = {};

  for (int k0 = 0; k0 < K; k0 += 64) {
    // Stage A (64 rows m x 64 cols k) and B (64 rows k x 64 cols n) tiles.
    // Identical 8-element chunk mapping: 512 chunks each, 4 per thread.
#pragma unroll
    for (int c = tid; c < 512; c += 128) {
      int r = c >> 3, cc = (c & 7) << 3;
      stage8<TA>(&As[r * 64 + cc], &A[(size_t)(m0 + r) * K + k0 + cc]);
      stage8<TB>(&Bs[r * 64 + cc], &Bm[(size_t)(k0 + r) * Nc + n0 + cc]);
    }
    async_wait0();
    __syncthreads();

    // Preload all fragments, then issue WMMAs back-to-back (distinct accs).
    v16bf a0 = lds_a_16x32(&As[wv * 16 * 64], 64, lane);       // k = k0..k0+31
    v16bf a1 = lds_a_16x32(&As[wv * 16 * 64 + 32], 64, lane);  // k = k0+32..+63
    v16bf bl[4], bh[4];
#pragma unroll
    for (int t = 0; t < 4; ++t) {
      bl[t] = lds_b_32x16(&Bs[t * 16], 64, lane);            // k rows 0..31
      bh[t] = lds_b_32x16(&Bs[32 * 64 + t * 16], 64, lane);  // k rows 32..63
    }
#pragma unroll
    for (int t = 0; t < 4; ++t) acc[t] = wmma_bf16(a0, bl[t], acc[t]);
#pragma unroll
    for (int t = 0; t < 4; ++t) acc[t] = wmma_bf16(a1, bh[t], acc[t]);
    __syncthreads();
  }

  // Store C: lane holds (m = mb+j, n = nl) per 16-col chunk t.
#pragma unroll
  for (int t = 0; t < 4; ++t) {
#pragma unroll
    for (int j = 0; j < 8; ++j) {
      int    gm  = m0 + wv * 16 + mb + j;
      int    gn  = n0 + t * 16 + nl;
      size_t idx = (size_t)gm * Nc + gn;
      if (BF16_OUT) {
        ((__bf16*)Cv)[idx] = (__bf16)acc[t][j];
      } else {
        ((float*)Cv)[idx] = acc[t][j] + bias[gn];
      }
    }
  }
}

// ---------------------------------------------------------------------------
// Flash attention with additive position bias + causal mask.
// Block = 128 threads (4 waves) = one (b, h, 64-query-row) tile.
// Per KV tile of 32: 4 WMMAs for S(16x32), f32 online softmax, 4 WMMAs for
// O += P*V. Bias loads are unconditional (always in-row-bounds); the causal
// mask is applied with selects so EXEC stays all-ones around the WMMAs.
// ---------------------------------------------------------------------------
__global__ __launch_bounds__(128) void flash_attn_kernel(
    const __bf16* __restrict__ qkv,      // [B*N][3*DIM] bf16
    const float* __restrict__ pos_bias,  // [H][N][N] f32
    __bf16* __restrict__ attnb) {        // [B*N][DIM] bf16
  __shared__ __align__(16) __bf16 Kt[32 * 64];
  __shared__ __align__(16) __bf16 Vt[32 * 64];
  __shared__ __align__(16) __bf16 Ps[4][16 * 32];

  const int tid   = threadIdx.x;
  const int wv    = tid >> 5;
  const int lane  = tid & 31;
  const int nl    = lane & 15;
  const int half  = lane >> 4;
  const int mb    = half << 3;   // C-layout row base (0 or 8)
  const int kb8   = half << 3;   // A-layout k base (0 or 8)
  const int qbase = blockIdx.x * 64;
  const int h     = blockIdx.y;
  const int b     = blockIdx.z;
  const size_t rowbase = (size_t)b * SEQ;

  // --- Q fragments for this wave (rows qbase + wv*16 .. +15) ---------------
  const int qm_a = qbase + wv * 16 + nl;  // A-frag: m = lane & 15
  const __bf16* qrow = qkv + (rowbase + qm_a) * D3 + h * HD;
  v16bf qa0, qa1;
#pragma unroll
  for (int j = 0; j < 8; ++j) {
    qa0[j]     = qrow[kb8 + j];
    qa0[j + 8] = qrow[16 + kb8 + j];
    qa1[j]     = qrow[32 + kb8 + j];
    qa1[j + 8] = qrow[48 + kb8 + j];
  }

  float rowmax[8], rowsum[8];
#pragma unroll
  for (int j = 0; j < 8; ++j) { rowmax[j] = NEGINF; rowsum[j] = 0.0f; }
  v8f o0 = {}, o1 = {}, o2 = {}, o3 = {};

  const int qmrow = qbase + wv * 16 + mb;  // C-layout rows qmrow + j
  const int jmax  = qbase + 63;            // causal: last needed key index
  // Per-lane bias base: row (qmrow+j), column (j0 + nl [+16])
  const float* pbias = pos_bias + ((size_t)h * SEQ + qmrow) * SEQ + nl;

  for (int j0 = 0; j0 <= jmax; j0 += 32) {
    // ---- Stage K/V tiles (32 keys x 64 dims bf16) via async LDS copies ----
#pragma unroll
    for (int c = tid; c < 256; c += 128) {
      int r = c >> 3, cc = (c & 7) << 3;
      const __bf16* base = qkv + (rowbase + j0 + r) * D3 + h * HD;
      g2l_copy16(&Kt[r * 64 + cc], base + DIM + cc);
      g2l_copy16(&Vt[r * 64 + cc], base + 2 * DIM + cc);
    }
    async_wait0();
    __syncthreads();

    // ---- S = Q * K^T : preload all 4 K fragments, then 4 WMMAs ------------
    v16bf kf00 = lds_bT_16rows(&Kt[0 * 64 + 0],   64, lane);
    v16bf kf01 = lds_bT_16rows(&Kt[0 * 64 + 32],  64, lane);
    v16bf kf10 = lds_bT_16rows(&Kt[16 * 64 + 0],  64, lane);
    v16bf kf11 = lds_bT_16rows(&Kt[16 * 64 + 32], 64, lane);
    v8f s0 = {}, s1 = {};
    s0 = wmma_bf16(qa0, kf00, s0);
    s1 = wmma_bf16(qa0, kf10, s1);
    s0 = wmma_bf16(qa1, kf01, s0);
    s1 = wmma_bf16(qa1, kf11, s1);

    // ---- Unconditional bias loads (always in-bounds within the row) -------
    float b0[8], b1[8];
#pragma unroll
    for (int j = 0; j < 8; ++j) {
      const float* brow = pbias + (size_t)j * SEQ + j0;
      b0[j] = brow[0];
      b1[j] = brow[16];
    }

    // ---- scale + bias + causal mask via selects (no EXEC divergence) ------
    float tm[8], p0[8], p1[8];
    const int c0 = j0 + nl, c1 = j0 + 16 + nl;
#pragma unroll
    for (int j = 0; j < 8; ++j) {
      int   m  = qmrow + j;
      float v0 = s0[j] * SCALE + b0[j];
      float v1 = s1[j] * SCALE + b1[j];
      v0 = (c0 <= m) ? v0 : NEGINF;
      v1 = (c1 <= m) ? v1 : NEGINF;
      s0[j] = v0; s1[j] = v1;
      tm[j] = fmaxf(v0, v1);
    }
    // row-max across the 16 lanes of this row group
#pragma unroll
    for (int off = 1; off < 16; off <<= 1) {
#pragma unroll
      for (int j = 0; j < 8; ++j) tm[j] = fmaxf(tm[j], __shfl_xor(tm[j], off, 32));
    }

    float corr[8], psum[8];
#pragma unroll
    for (int j = 0; j < 8; ++j) {
      float nm  = fmaxf(rowmax[j], tm[j]);
      corr[j]   = __expf(rowmax[j] - nm);
      rowmax[j] = nm;
      p0[j]   = __expf(s0[j] - nm);
      p1[j]   = __expf(s1[j] - nm);
      psum[j] = p0[j] + p1[j];
    }
#pragma unroll
    for (int off = 1; off < 16; off <<= 1) {
#pragma unroll
      for (int j = 0; j < 8; ++j) psum[j] += __shfl_xor(psum[j], off, 32);
    }
#pragma unroll
    for (int j = 0; j < 8; ++j) {
      rowsum[j] = rowsum[j] * corr[j] + psum[j];
      o0[j] *= corr[j]; o1[j] *= corr[j]; o2[j] *= corr[j]; o3[j] *= corr[j];
    }

    // ---- P (16x32 f32) -> bf16 via LDS, re-read in A-fragment layout ------
    __bf16* pw = &Ps[wv][0];
#pragma unroll
    for (int j = 0; j < 8; ++j) {
      pw[(mb + j) * 32 + nl]      = (__bf16)p0[j];
      pw[(mb + j) * 32 + 16 + nl] = (__bf16)p1[j];
    }
    __syncthreads();  // P visible; also fences K/V reuse below

    // ---- O += P * V : preload P + all 4 V fragments, then 4 WMMAs ---------
    v16bf pa  = lds_a_16x32(pw, 32, lane);
    v16bf vf0 = lds_b_32x16(&Vt[0],  64, lane);
    v16bf vf1 = lds_b_32x16(&Vt[16], 64, lane);
    v16bf vf2 = lds_b_32x16(&Vt[32], 64, lane);
    v16bf vf3 = lds_b_32x16(&Vt[48], 64, lane);
    o0 = wmma_bf16(pa, vf0, o0);
    o1 = wmma_bf16(pa, vf1, o1);
    o2 = wmma_bf16(pa, vf2, o2);
    o3 = wmma_bf16(pa, vf3, o3);
    __syncthreads();  // protect Kt/Vt before the next tile load

    // Prefetch next pos_bias tile (global_prefetch_b8)
    if (j0 + 32 <= jmax) {
      __builtin_prefetch(pbias + (size_t)(j0 + 32), 0, 1);
    }
  }

  // ---- Normalize and store O (bf16, [B*N][DIM], col = h*64 + t*16 + nl) ---
#pragma unroll
  for (int j = 0; j < 8; ++j) {
    float  inv  = 1.0f / rowsum[j];
    size_t base = (rowbase + qmrow + j) * DIM + h * HD + nl;
    attnb[base +  0] = (__bf16)(o0[j] * inv);
    attnb[base + 16] = (__bf16)(o1[j] * inv);
    attnb[base + 32] = (__bf16)(o2[j] * inv);
    attnb[base + 48] = (__bf16)(o3[j] * inv);
  }
}

// ---------------------------------------------------------------------------
// Launch: QKV GEMM (f32 in, bf16 out) -> flash attention -> projection GEMM
// ---------------------------------------------------------------------------
extern "C" void kernel_launch(void* const* d_in, const int* in_sizes, int n_in,
                              void* d_out, int out_size, void* d_ws, size_t ws_size,
                              hipStream_t stream) {
  const float* x        = (const float*)d_in[0];  // [B,N,D]
  const float* pos_bias = (const float*)d_in[1];  // [H,N,N]
  const float* w_qkv    = (const float*)d_in[2];  // [D,3D]
  const float* w_proj   = (const float*)d_in[3];  // [D,D]
  const float* b_proj   = (const float*)d_in[4];  // [D]
  float*       out      = (float*)d_out;          // [B,N,D]

  // Workspace: qkv (bf16, 48MB) + attn (bf16, 16MB)
  constexpr size_t QKVE = (size_t)MROWS * D3;
  __bf16* qkvb  = (__bf16*)d_ws;
  __bf16* attnb = qkvb + QKVE;

  // 1) QKV projection: [8192,1024](f32) x [1024,3072](f32) -> bf16
  gemm64_kernel<float, float, true><<<dim3(D3 / 64, MROWS / 64), 128, 0, stream>>>(
      x, w_qkv, (void*)qkvb, nullptr, DIM, D3);

  // 2) Flash attention per (b, h, 64 query rows)
  flash_attn_kernel<<<dim3(SEQ / 64, HEADS, BATCH), 128, 0, stream>>>(
      qkvb, pos_bias, attnb);

  // 3) Output projection: [8192,1024](bf16) x [1024,1024](f32) + bias -> f32
  gemm64_kernel<__bf16, float, false><<<dim3(DIM / 64, MROWS / 64), 128, 0, stream>>>(
      attnb, w_proj, (void*)out, b_proj, DIM, DIM);
}
